// HardwareAwareSNN_26173530702576
// MI455X (gfx1250) — compile-verified, matrix operand order checked by reading, not run on hardware
//
#include <hip/hip_runtime.h>
#include <math.h>
#include <stdint.h>

typedef _Float16 h16;
typedef __attribute__((ext_vector_type(16))) _Float16 v16h;
typedef __attribute__((ext_vector_type(8)))  _Float16 v8h;
typedef __attribute__((ext_vector_type(8)))  float    v8f;

#define T_STEPS 25
#define Bsz 4096
#define Din 784
#define Hsz 500
#define Osz 10
#define KP1 800    // D padded to multiple of 32 (WMMA K-step)
#define NP1 512    // H padded to multiple of 16
#define KP2 512    // GEMM2 K (= NP1)
#define NP2 16     // O padded to one 16-wide N tile
#define BETA 0.9f
#define THR  1.0f
#define NSTD 0.02f

#define SALT_SP 0x1234567855667788ull
#define SALT_N1 0xA5A5A5A5DEADBEEFull
#define SALT_N2 0x0F0F0F0FCAFEBABEull

// LDS pitches chosen so 16 lanes' b128 reads start at 16 distinct bank-groups:
// pitch 40 h16 = 20 dwords; 20*row mod 64 covers all multiples of 4 (5 coprime 16).
#define BP1 40
// pitch 520 h16 = 260 dwords; 260 mod 64 = 4 -> row*4 distinct multiples of 4.
#define WP2 520

// ---- CDNA5 async global->LDS staging via inline asm (ASYNCcnt path) ----
// ISA 15.18.3: GLOBAL_LOAD_ASYNC_TO_LDS_B128, GV mode: vdst = LDS byte addr VGPR,
// vaddr = 64-bit global addr, SADDR = off. Tracked by ASYNCcnt (s_wait_asynccnt).
// LDS byte address = low 32 bits of the generic pointer (ISA 10.2 aperture map).
__device__ __forceinline__ void async_copy_b128(const h16* g, h16* l) {
  const uint32_t loff  = (uint32_t)(uintptr_t)(void*)l;
  const uint64_t gaddr = (uint64_t)(uintptr_t)(const void*)g;
  asm volatile("global_load_async_to_lds_b128 %0, %1, off"
               :
               : "v"(loff), "v"(gaddr)
               : "memory");
}
__device__ __forceinline__ void wait_async0() {
  asm volatile("s_wait_asynccnt 0x0" ::: "memory");
}

// ---------------- counter-based RNG (stateless, deterministic) ----------------
__device__ __forceinline__ uint32_t mix64(uint64_t z) {
  z += 0x9E3779B97F4A7C15ull;
  z = (z ^ (z >> 30)) * 0xBF58476D1CE4E5B9ull;
  z = (z ^ (z >> 27)) * 0x94D049BB133111EBull;
  z ^= (z >> 31);
  return (uint32_t)(z ^ (z >> 32));
}
__device__ __forceinline__ float u01(uint32_t h) {        // [0,1)
  return (float)(h >> 8) * (1.0f / 16777216.0f);
}
__device__ __forceinline__ float gauss(uint64_t idx, uint64_t salt) {
  uint32_t ha = mix64(idx * 2ull + salt);
  uint32_t hb = mix64(idx * 2ull + 1ull + salt);
  float u1 = ((float)ha + 1.0f) * (1.0f / 4294967296.0f); // (0,1]
  float u2 = (float)hb * (1.0f / 4294967296.0f);
  float r = sqrtf(-2.0f * __logf(u1));
  return r * __cosf(6.28318530717958647f * u2);
}

// Build a WMMA 16-bit fragment from two 16B chunks (ISA 7.12.2 16x32 layout):
// lane L -> row (L&15), half (L>>4); v0..v3 <- K=half*8+0..7 ; v4..v7 <- K=16+half*8+0..7
__device__ __forceinline__ v16h pack_frag(v8h lo, v8h hi) {
  v16h a;
#pragma unroll
  for (int i = 0; i < 8; ++i) { a[i] = lo[i]; a[i + 8] = hi[i]; }
  return a;
}

// ---------------- state init: zero membrane potentials ----------------
__global__ void snn_init(float* __restrict__ mem1, float* __restrict__ mem2) {
  const int n1 = Bsz * NP1, n2 = Bsz * NP2;
  int i = blockIdx.x * blockDim.x + threadIdx.x;
  if (i < n1) mem1[i] = 0.0f;
  if (i < n2) mem2[i] = 0.0f;
}

// ---------------- Bernoulli rate-coded spikes, fp16, padded [Bsz][KP1] ----------------
__global__ void snn_spikes(const float* __restrict__ x, h16* __restrict__ S, int t) {
  int idx = blockIdx.x * blockDim.x + threadIdx.x;      // one 8-elem chunk each
  const int chunks = KP1 / 8;
  if (idx >= Bsz * chunks) return;
  const int b  = idx / chunks;
  const int kc = (idx % chunks) * 8;
  v8h v;
#pragma unroll
  for (int i = 0; i < 8; ++i) {
    const int k = kc + i;
    float s = 0.0f;
    if (k < Din) {
      const uint64_t fi = ((uint64_t)t * Bsz + b) * (uint64_t)Din + k;
      const float u = u01(mix64(fi * 2ull + SALT_SP));
      s = (u < x[(size_t)b * Din + k]) ? 1.0f : 0.0f;
    }
    v[i] = (h16)s;
  }
  *(v8h*)(S + (size_t)b * KP1 + kc) = v;
}

// ---------------- noisy weight read: W1nT[n][k] = W1[k][n] + N(0, NSTD) ----------------
__global__ void snn_w1(const float* __restrict__ W1, h16* __restrict__ W1nT, int t) {
  int idx = blockIdx.x * blockDim.x + threadIdx.x;      // over NP1*KP1
  if (idx >= NP1 * KP1) return;
  const int n = idx / KP1;
  const int k = idx % KP1;
  float w = 0.0f;
  if (n < Hsz && k < Din) {
    const uint64_t fi = ((uint64_t)t * Din + k) * (uint64_t)Hsz + n;  // n1[t,d,h]
    w = W1[(size_t)k * Hsz + n] + NSTD * gauss(fi, SALT_N1);
  }
  W1nT[idx] = (h16)w;
}

__global__ void snn_w2(const float* __restrict__ W2, h16* __restrict__ W2nT, int t) {
  int idx = blockIdx.x * blockDim.x + threadIdx.x;      // over NP2*KP2
  if (idx >= NP2 * KP2) return;
  const int o = idx / KP2;
  const int h = idx % KP2;
  float w = 0.0f;
  if (o < Osz && h < Hsz) {
    const uint64_t fi = ((uint64_t)t * Hsz + h) * (uint64_t)Osz + o;  // n2[t,h,o]
    w = W2[(size_t)h * Osz + o] + NSTD * gauss(fi, SALT_N2);
  }
  W2nT[idx] = (h16)w;
}

// ---------------- GEMM1 (WMMA f16->f32) fused with layer-1 LIF ----------------
// Block = 8 waves = 128(M) x 64(N); B panel double-buffered in LDS, shared by all waves.
// Panel for step k+1 is staged asynchronously (ASYNCcnt) while step k's WMMAs run.
__global__ void __launch_bounds__(256)
snn_gemm1_lif(const h16* __restrict__ S, const h16* __restrict__ W1nT,
              const float* __restrict__ b1, float* __restrict__ mem1,
              h16* __restrict__ spk1) {
  __shared__ h16 Bs[2][64 * BP1];

  const int tid  = threadIdx.x;
  const int lane = tid & 31;
  const int wid  = tid >> 5;                   // 0..7 -> M-tile within block
  const int bm   = blockIdx.x >> 3;            // 0..31
  const int bn   = blockIdx.x & 7;             // 0..7
  const int m0   = (bm * 8 + wid) * 16;
  const int n0   = bn * 64;

  // per-lane A base pointer (row = m0 + lane&15, first chunk at half*8)
  const h16* pa = S + (size_t)(m0 + (lane & 15)) * KP1 + (lane >> 4) * 8;

  // this thread's slice of the 64x32 B panel (one b128 per thread)
  const int srow = tid >> 2;                   // 0..63
  const int sch  = tid & 3;                    // 0..3 (8 h16 each)
  const h16* gsrc = W1nT + (size_t)(n0 + srow) * KP1 + sch * 8;
  h16* ldst0 = &Bs[0][srow * BP1 + sch * 8];
  h16* ldst1 = &Bs[1][srow * BP1 + sch * 8];

  // prologue: stage panel 0 into buffer 0
  async_copy_b128(gsrc, ldst0);
  wait_async0();
  __syncthreads();

  v8f acc[4] = {};
  const int NT = KP1 / 32;                     // 25
  const int nl = lane & 15, hh = lane >> 4;
#pragma unroll 1
  for (int kt = 0; kt < NT; ++kt) {
    const int cur = kt & 1;
    h16* nxt = cur ? ldst0 : ldst1;

    // kick off next panel before compute (latency hidden behind WMMAs)
    if (kt + 1 < NT) async_copy_b128(gsrc + (kt + 1) * 32, nxt);

    const v16h a = pack_frag(*(const v8h*)(pa + kt * 32),
                             *(const v8h*)(pa + kt * 32 + 16));
    const h16* cbuf = &Bs[cur][0];
    v16h bf[4];
#pragma unroll
    for (int j = 0; j < 4; ++j) {              // batch all 8 ds_load_b128 first
      const h16* p = cbuf + (j * 16 + nl) * BP1 + hh * 8;
      bf[j] = pack_frag(*(const v8h*)p, *(const v8h*)(p + 16));
    }
#pragma unroll
    for (int j = 0; j < 4; ++j) {
      acc[j] = __builtin_amdgcn_wmma_f32_16x16x32_f16(
          false, a, false, bf[j], (short)0, acc[j], false, false);
    }

    wait_async0();                             // next panel resident in LDS
    __syncthreads();
  }

  // C layout: vgpr r, lane L -> (m = r + 8*(L>>4), n = L&15)
#pragma unroll
  for (int j = 0; j < 4; ++j) {
    const int n = n0 + j * 16 + nl;
    const float bias = (n < Hsz) ? b1[n] : 0.0f;
#pragma unroll
    for (int r = 0; r < 8; ++r) {
      const int m = m0 + r + 8 * hh;
      const size_t o = (size_t)m * NP1 + n;
      const float old = mem1[o];
      const float cur1 = acc[j][r] + bias;
      const float rst = (old - THR > 0.0f) ? THR : 0.0f;   // reset from *previous* mem
      const float nm  = BETA * old + cur1 - rst;
      mem1[o] = nm;
      spk1[o] = (h16)((nm - THR > 0.0f) ? 1.0f : 0.0f);
    }
  }
}

// ---------------- GEMM2 (WMMA) fused with layer-2 LIF + output record ----------------
// Block = 8 waves (8 M-tiles); full 16x512 W2nT staged once in LDS, reused 16 k-steps.
__global__ void __launch_bounds__(256)
snn_gemm2_lif(const h16* __restrict__ spk1, const h16* __restrict__ W2nT,
              const float* __restrict__ b2, float* __restrict__ mem2,
              float* __restrict__ out_spk, float* __restrict__ out_mem) {
  __shared__ h16 Ws[16 * WP2];

  const int tid  = threadIdx.x;
  const int lane = tid & 31;
  const int wid  = tid >> 5;
  const int m0   = (blockIdx.x * 8 + wid) * 16;

  // cooperative async stage of entire B (16 rows x 512 k), 4 b128 chunks per thread
#pragma unroll
  for (int c = 0; c < 4; ++c) {
    const int cid  = tid + c * 256;            // 0..1023 chunks of 8 h16
    const int row  = cid >> 6;                 // 512/8 = 64 chunks per row
    const int col8 = (cid & 63) * 8;
    async_copy_b128(W2nT + (size_t)row * KP2 + col8, &Ws[row * WP2 + col8]);
  }
  wait_async0();
  __syncthreads();

  const int nl = lane & 15, hh = lane >> 4;
  const h16* pa = spk1 + (size_t)(m0 + nl) * KP2 + hh * 8;

  v8f acc = {};
#pragma unroll 1
  for (int kt = 0; kt < KP2 / 32; ++kt) {
    const v16h a = pack_frag(*(const v8h*)(pa + kt * 32),
                             *(const v8h*)(pa + kt * 32 + 16));
    const h16* p = &Ws[nl * WP2 + kt * 32 + hh * 8];
    const v16h b = pack_frag(*(const v8h*)p, *(const v8h*)(p + 16));
    acc = __builtin_amdgcn_wmma_f32_16x16x32_f16(
        false, a, false, b, (short)0, acc, false, false);
  }

  const int n = nl;
  const float bias = (n < Osz) ? b2[n] : 0.0f;
#pragma unroll
  for (int r = 0; r < 8; ++r) {
    const int m = m0 + r + 8 * hh;
    const size_t o = (size_t)m * NP2 + n;
    const float old = mem2[o];
    const float cur = acc[r] + bias;
    const float rst = (old - THR > 0.0f) ? THR : 0.0f;
    const float nm  = BETA * old + cur - rst;
    mem2[o] = nm;
    if (n < Osz) {
      const size_t oo = (size_t)m * Osz + n;
      out_spk[oo] = (nm - THR > 0.0f) ? 1.0f : 0.0f;
      out_mem[oo] = nm;
    }
  }
}

// ---------------- host side ----------------
extern "C" void kernel_launch(void* const* d_in, const int* in_sizes, int n_in,
                              void* d_out, int out_size, void* d_ws, size_t ws_size,
                              hipStream_t stream) {
  (void)in_sizes; (void)n_in; (void)out_size; (void)ws_size;
  const float* x  = (const float*)d_in[0];
  const float* W1 = (const float*)d_in[1];
  const float* b1 = (const float*)d_in[2];
  const float* W2 = (const float*)d_in[3];
  const float* b2 = (const float*)d_in[4];
  float* out = (float*)d_out;

  // workspace carve-up (256B aligned)
  char* ws = (char*)d_ws;
  size_t off = 0;
  auto carve = [&](size_t bytes) -> void* {
    void* p = ws + off;
    off = (off + bytes + 255) & ~(size_t)255;
    return p;
  };
  h16*   S     = (h16*)  carve((size_t)Bsz * KP1 * sizeof(h16));   // spikes, padded
  h16*   W1nT  = (h16*)  carve((size_t)NP1 * KP1 * sizeof(h16));   // noisy W1^T
  h16*   W2nT  = (h16*)  carve((size_t)NP2 * KP2 * sizeof(h16));   // noisy W2^T
  h16*   spk1  = (h16*)  carve((size_t)Bsz * NP1 * sizeof(h16));   // layer-1 spikes
  float* mem1  = (float*)carve((size_t)Bsz * NP1 * sizeof(float));
  float* mem2  = (float*)carve((size_t)Bsz * NP2 * sizeof(float));

  // zero membrane state (harness does not re-poison between replays)
  {
    const int n = Bsz * NP1;          // >= Bsz*NP2
    snn_init<<<(n + 255) / 256, 256, 0, stream>>>(mem1, mem2);
  }

  const int spkBlocks = (Bsz * (KP1 / 8) + 255) / 256;   // 1600
  const int w1Blocks  = (NP1 * KP1 + 255) / 256;         // 1600
  const int w2Blocks  = (NP2 * KP2 + 255) / 256;         // 32
  const int g1Blocks  = 32 * 8;                          // (4096/128) M-blocks x 8 N-strips
  const int g2Blocks  = 4096 / 128;                      // 8 M-tiles per block

  for (int t = 0; t < T_STEPS; ++t) {
    snn_spikes<<<spkBlocks, 256, 0, stream>>>(x, S, t);
    snn_w1<<<w1Blocks, 256, 0, stream>>>(W1, W1nT, t);
    snn_w2<<<w2Blocks, 256, 0, stream>>>(W2, W2nT, t);
    snn_gemm1_lif<<<g1Blocks, 256, 0, stream>>>(S, W1nT, b1, mem1, spk1);
    snn_gemm2_lif<<<g2Blocks, 256, 0, stream>>>(
        spk1, W2nT, b2, mem2,
        out + (size_t)t * Bsz * Osz,
        out + ((size_t)T_STEPS + t) * Bsz * Osz);
  }
}
